// LinRec_Model_34299608826449
// MI455X (gfx1250) — matrix operand order repaired and checked by reading, not compiled, not run on hardware
//
#include <hip/hip_runtime.h>
#include <math.h>
#include <stdint.h>

typedef __bf16 bf16_t;
typedef __attribute__((ext_vector_type(16))) __bf16 v16bf;
typedef __attribute__((ext_vector_type(8)))  __bf16 v8bf;
typedef __attribute__((ext_vector_type(8)))  float  v8f;
typedef __attribute__((ext_vector_type(4)))  int    v4i;

#define D_EMB  256
#define SEQLEN 200
#define BATCH_ 256
#define MTOK   (SEQLEN*BATCH_)   /* 51200 tokens, layout [L,B,D], tok = l*256+b */
#define NHEAD  8
#define DH     32

#if defined(__has_builtin)
#  if __has_builtin(__builtin_amdgcn_global_load_async_to_lds_b128)
#    define USE_ASYNC_LDS 1
#  endif
#endif
#ifndef USE_ASYNC_LDS
#  define USE_ASYNC_LDS 0
#endif

typedef __attribute__((address_space(1))) v4i gv4i_t;   // global 16B chunk
typedef __attribute__((address_space(3))) v4i lv4i_t;   // LDS 16B chunk

static __device__ __forceinline__ v8f wmma_bf16(v16bf a, v16bf b, v8f c) {
  // D = A(16x32) * B(32x16) + C, f32 accumulate
  return __builtin_amdgcn_wmma_f32_16x16x32_bf16(false, a, false, b, (short)0, c, false, false);
}

static __device__ __forceinline__ float wave_red_add(float x) {
#pragma unroll
  for (int m = 16; m > 0; m >>= 1) x += __shfl_xor(x, m, 32);
  return x;
}

// ---------------------------------------------------------------------------
// Weight prep: W[k][n] f32 -> Wt[n][k] bf16 (so B-fragments load contiguously)
// ---------------------------------------------------------------------------
__global__ void prep_w(const float* __restrict__ W, bf16_t* __restrict__ Wt) {
  int idx = blockIdx.x * 256 + threadIdx.x;   // 65536 elements, grid=256
  int k = idx >> 8, n = idx & 255;
  Wt[n * D_EMB + k] = (bf16_t)W[idx];
}

// ---------------------------------------------------------------------------
// Embedding: x[l*B+b, :] = (item_emb[seq]*sqrt(D) + pos_emb[pos]) * mask
// Writes f32 (residual path) + bf16 (GEMM A path).
// ---------------------------------------------------------------------------
__global__ void embed_kernel(const int* __restrict__ seq, const int* __restrict__ pos,
                             const float* __restrict__ item_emb,
                             const float* __restrict__ pos_emb,
                             float* __restrict__ X, bf16_t* __restrict__ Xb) {
  int t = blockIdx.x;            // token
  int l = t >> 8, b = t & 255;
  int si = seq[b * SEQLEN + l];
  int pi = pos[b * SEQLEN + l];
  int c = threadIdx.x;
  float v = item_emb[(size_t)si * D_EMB + c] * 16.0f + pos_emb[(size_t)pi * D_EMB + c];
  if (si == 0) v = 0.0f;
  X [(size_t)t * D_EMB + c] = v;
  Xb[(size_t)t * D_EMB + c] = (bf16_t)v;
}

// ---------------------------------------------------------------------------
// WMMA GEMM: C[M,256] = A_bf16[M,256] @ W[256,256] + bias, fused epilogues.
// grid = (M/256, 4). 512 threads = 16 waves; wave w owns rows [w*16, w*16+16)
// of the 256-row slab and computes 16 x 64 (4 WMMA N-tiles). The 64x256 B
// panel (32KB, contiguous rows of Wt) is staged in LDS (async if available).
// EPI: 0 = +bias -> f32
//      1 = +bias + R -> f32
//      2 = relu(+bias) -> bf16
//      3 = +bias + R, mask(seq) -> f32 + bf16
//      4 = +bias -> bf16
// ---------------------------------------------------------------------------
template<int EPI>
__global__ __launch_bounds__(512)
void gemm256(const bf16_t* __restrict__ A, const bf16_t* __restrict__ Wt,
             const float* __restrict__ bias, const float* __restrict__ R,
             const int* __restrict__ seq,
             float* __restrict__ Cf, bf16_t* __restrict__ Cb) {
  __shared__ bf16_t Bs[64 * 256];            // 32 KB B panel
  const int ncol0 = blockIdx.y * 64;

  {
    const bf16_t* src = Wt + (size_t)ncol0 * D_EMB;   // 64 contiguous rows
#if USE_ASYNC_LDS
#pragma unroll
    for (int j = 0; j < 4; ++j) {
      int idx = threadIdx.x + j * 512;                // 16B chunk id (2048 total)
      gv4i_t* g = (gv4i_t*)(unsigned long long)(uintptr_t)(src + (size_t)idx * 8);
      lv4i_t* l = (lv4i_t*)(unsigned int)(uintptr_t)(&Bs[(size_t)idx * 8]);
      __builtin_amdgcn_global_load_async_to_lds_b128(g, l, 0, 0);
    }
#  if defined(__has_builtin) && __has_builtin(__builtin_amdgcn_s_wait_asynccnt)
    __builtin_amdgcn_s_wait_asynccnt(0);
#  else
    asm volatile("s_wait_asynccnt 0" ::: "memory");
#  endif
#else
#pragma unroll
    for (int j = 0; j < 4; ++j) {
      int idx = threadIdx.x + j * 512;
      ((v8bf*)Bs)[idx] = ((const v8bf*)src)[idx];
    }
#endif
    __syncthreads();
  }

  const int w    = threadIdx.x >> 5;
  const int lane = threadIdx.x & 31;
  const int l16  = lane & 15;
  const int hi   = lane >> 4;
  const int rowbase = blockIdx.x * 256 + w * 16;
  const int row     = rowbase + l16;

  const bf16_t* ap  = A  + (size_t)row * D_EMB + hi * 16;  // A-frag K base
  const bf16_t* bsl = Bs + (size_t)l16 * D_EMB + hi * 16;  // B-frag base in LDS

  v8f acc[4] = {v8f{}, v8f{}, v8f{}, v8f{}};

#pragma unroll
  for (int ks = 0; ks < 8; ++ks) {
    v8bf a0 = *(const v8bf*)(ap + ks * 32);
    v8bf a1 = *(const v8bf*)(ap + ks * 32 + 8);
    v16bf af;
#pragma unroll
    for (int j = 0; j < 8; ++j) { af[j] = a0[j]; af[8 + j] = a1[j]; }
#pragma unroll
    for (int n = 0; n < 4; ++n) {
      v16bf bf = *(const v16bf*)(bsl + n * 16 * D_EMB + ks * 32);
      acc[n] = wmma_bf16(af, bf, acc[n]);
    }
  }

  float bb[4];
#pragma unroll
  for (int n = 0; n < 4; ++n) bb[n] = bias[ncol0 + n * 16 + l16];

#pragma unroll
  for (int r = 0; r < 8; ++r) {
    int orow = rowbase + r + 8 * hi;
    float msk = 1.0f;
    if (EPI == 3) {
      int b = orow & 255, l = orow >> 8;
      if (seq[b * SEQLEN + l] == 0) msk = 0.0f;
    }
#pragma unroll
    for (int n = 0; n < 4; ++n) {
      int col = ncol0 + n * 16 + l16;
      float v = acc[n][r] + bb[n];
      if (EPI == 1 || EPI == 3) v += R[(size_t)orow * D_EMB + col];
      if (EPI == 2) v = fmaxf(v, 0.0f);
      if (EPI == 3) v *= msk;
      if (EPI == 0 || EPI == 1 || EPI == 3) Cf[(size_t)orow * D_EMB + col] = v;
      if (EPI == 2 || EPI == 3 || EPI == 4) Cb[(size_t)orow * D_EMB + col] = (bf16_t)v;
    }
  }
}

// ---------------------------------------------------------------------------
// ELU + L2-normalize over dh=32 (one lane per element, 32-lane groups == heads)
// ---------------------------------------------------------------------------
__global__ void elunorm_kernel(const float* __restrict__ Q, const float* __restrict__ K,
                               bf16_t* __restrict__ Qn, bf16_t* __restrict__ Kn) {
  size_t e = (size_t)blockIdx.x * 256 + threadIdx.x;
  float q = Q[e], k = K[e];
  float eq = (q > 0.f) ? q : (__expf(q) - 1.f);
  float ek = (k > 0.f) ? k : (__expf(k) - 1.f);
  float sq = wave_red_add(eq * eq);
  float sk = wave_red_add(ek * ek);
  Qn[e] = (bf16_t)(eq * rsqrtf(sq));
  Kn[e] = (bf16_t)(ek * rsqrtf(sk));
}

// ---------------------------------------------------------------------------
// Linear attention per (l,h): KV(32x32) = Kn^T @ V  (K=256 batch), then
// ctx = Qn @ KV / sqrt(dh) -> bf16.  grid = L*H = 1600, 512 threads = 16 waves.
// ---------------------------------------------------------------------------
__global__ __launch_bounds__(512)
void attn_kernel(const bf16_t* __restrict__ Qn, const bf16_t* __restrict__ Kn,
                 const bf16_t* __restrict__ Vb, bf16_t* __restrict__ ctxb) {
  __shared__ bf16_t kvT[32][32];         // KV stored transposed: kvT[j][i]=KV[i][j]
  const int w    = threadIdx.x >> 5;
  const int lane = threadIdx.x & 31;
  const int l16  = lane & 15;
  const int hi   = lane >> 4;
  const int l    = blockIdx.x >> 3;
  const int h    = blockIdx.x & 7;
  const int lB   = l * BATCH_;
  const int hoff = h * DH;

  // Phase 1: waves 0..3 build KV = Kmat^T(32x256) @ Vmat(256x32), 2x2 tiles
  if (w < 4) {
    const int tI = w >> 1, tJ = w & 1;
    v8f acc = {};
    for (int ks = 0; ks < 8; ++ks) {
      v16bf af, bf;
      int kb0 = ks * 32 + hi * 16;
#pragma unroll
      for (int j = 0; j < 8; ++j) {     // A[m,k] = Kn[tok=lB+k, hoff+tI*16+m]
        af[j]     = Kn[(size_t)(lB + kb0 + j)      * D_EMB + hoff + tI * 16 + l16];
        af[8 + j] = Kn[(size_t)(lB + kb0 + 8 + j)  * D_EMB + hoff + tI * 16 + l16];
      }
      int ksb = ks * 32 + hi * 16;
#pragma unroll
      for (int m = 0; m < 16; ++m)      // B[k,n] = Vb[tok=lB+k, hoff+tJ*16+n]
        bf[m] = Vb[(size_t)(lB + ksb + m) * D_EMB + hoff + tJ * 16 + l16];
      acc = wmma_bf16(af, bf, acc);
    }
#pragma unroll
    for (int r = 0; r < 8; ++r)
      kvT[tJ * 16 + l16][tI * 16 + r + 8 * hi] = (bf16_t)acc[r];
  }
  __syncthreads();

  // Phase 2: 16 waves, each does a 16-row slab of ctx = Qn(256x32) @ KV(32x32)
  const int row0 = lB + w * 16 + l16;
  const bf16_t* qp = Qn + (size_t)row0 * D_EMB + hoff + hi * 16;
  v8bf q0 = *(const v8bf*)qp;
  v8bf q1 = *(const v8bf*)(qp + 8);
  v16bf af;
#pragma unroll
  for (int j = 0; j < 8; ++j) { af[j] = q0[j]; af[8 + j] = q1[j]; }

#pragma unroll
  for (int n = 0; n < 2; ++n) {
    v16bf bf = *(const v16bf*)&kvT[n * 16 + l16][hi * 16];  // contiguous 32B in LDS
    v8f acc = {};
    acc = wmma_bf16(af, bf, acc);
#pragma unroll
    for (int r = 0; r < 8; ++r)
      ctxb[(size_t)(lB + w * 16 + r + 8 * hi) * D_EMB + hoff + n * 16 + l16] =
          (bf16_t)(acc[r] * 0.17677669529663687f);   // 1/sqrt(32)
  }
}

// ---------------------------------------------------------------------------
// Fused double LayerNorm per token: LN(.,g1,b1,1e-12) then LN(.,g2,b2,1e-8)
// 256 threads = 8 waves, one wave per token. Writes f32 + bf16.
// ---------------------------------------------------------------------------
__global__ void double_ln_kernel(const float* __restrict__ X,
                                 const float* __restrict__ g1, const float* __restrict__ b1,
                                 const float* __restrict__ g2, const float* __restrict__ b2,
                                 float* __restrict__ Y, bf16_t* __restrict__ Yb) {
  int tok  = blockIdx.x * 8 + (threadIdx.x >> 5);
  int lane = threadIdx.x & 31;
  const float* xp = X + (size_t)tok * D_EMB;
  float v[8];
#pragma unroll
  for (int j = 0; j < 8; ++j) v[j] = xp[lane + 32 * j];

  float s = 0.f;
#pragma unroll
  for (int j = 0; j < 8; ++j) s += v[j];
  float mu = wave_red_add(s) * (1.0f / 256.0f);
  float ss = 0.f;
#pragma unroll
  for (int j = 0; j < 8; ++j) { float d = v[j] - mu; ss += d * d; }
  float rinv = rsqrtf(wave_red_add(ss) * (1.0f / 256.0f) + 1e-12f);

  float y[8];
#pragma unroll
  for (int j = 0; j < 8; ++j) {
    int c = lane + 32 * j;
    y[j] = (v[j] - mu) * rinv * g1[c] + b1[c];
  }

  s = 0.f;
#pragma unroll
  for (int j = 0; j < 8; ++j) s += y[j];
  float mu2 = wave_red_add(s) * (1.0f / 256.0f);
  ss = 0.f;
#pragma unroll
  for (int j = 0; j < 8; ++j) { float d = y[j] - mu2; ss += d * d; }
  float rinv2 = rsqrtf(wave_red_add(ss) * (1.0f / 256.0f) + 1e-8f);

  float*  yp  = Y  + (size_t)tok * D_EMB;
  bf16_t* ypb = Yb + (size_t)tok * D_EMB;
#pragma unroll
  for (int j = 0; j < 8; ++j) {
    int c = lane + 32 * j;
    float o = (y[j] - mu2) * rinv2 * g2[c] + b2[c];
    yp[c]  = o;
    ypb[c] = (bf16_t)o;
  }
}

// ---------------------------------------------------------------------------
// Final LayerNorm (eps 1e-8) + transpose [L,B,D] -> [B,L,D]
// ---------------------------------------------------------------------------
__global__ void final_ln_kernel(const float* __restrict__ X,
                                const float* __restrict__ g, const float* __restrict__ b,
                                float* __restrict__ out) {
  int tok  = blockIdx.x * 8 + (threadIdx.x >> 5);
  int lane = threadIdx.x & 31;
  int l = tok >> 8, bb = tok & 255;
  const float* xp = X + (size_t)tok * D_EMB;
  float v[8];
#pragma unroll
  for (int j = 0; j < 8; ++j) v[j] = xp[lane + 32 * j];
  float s = 0.f;
#pragma unroll
  for (int j = 0; j < 8; ++j) s += v[j];
  float mu = wave_red_add(s) * (1.0f / 256.0f);
  float ss = 0.f;
#pragma unroll
  for (int j = 0; j < 8; ++j) { float d = v[j] - mu; ss += d * d; }
  float rinv = rsqrtf(wave_red_add(ss) * (1.0f / 256.0f) + 1e-8f);
  float* op = out + ((size_t)bb * SEQLEN + l) * D_EMB;
#pragma unroll
  for (int j = 0; j < 8; ++j) {
    int c = lane + 32 * j;
    op[c] = (v[j] - mu) * rinv * g[c] + b[c];
  }
}

// ---------------------------------------------------------------------------
extern "C" void kernel_launch(void* const* d_in, const int* in_sizes, int n_in,
                              void* d_out, int out_size, void* d_ws, size_t ws_size,
                              hipStream_t stream) {
  const int*   seq      = (const int*)d_in[0];
  const int*   pos      = (const int*)d_in[1];
  const float* item_emb = (const float*)d_in[2];
  const float* pos_emb  = (const float*)d_in[3];
  const float* Wq = (const float*)d_in[4];  const float* bq = (const float*)d_in[5];
  const float* Wk = (const float*)d_in[6];  const float* bk = (const float*)d_in[7];
  const float* Wv = (const float*)d_in[8];  const float* bv = (const float*)d_in[9];
  const float* Wd = (const float*)d_in[10]; const float* bd = (const float*)d_in[11];
  const float* ln1g = (const float*)d_in[12]; const float* ln1b = (const float*)d_in[13];
  const float* lnfg = (const float*)d_in[14]; const float* lnfb = (const float*)d_in[15];
  const float* W1 = (const float*)d_in[16]; const float* b1 = (const float*)d_in[17];
  const float* W2 = (const float*)d_in[18]; const float* b2 = (const float*)d_in[19];
  const float* lnLg = (const float*)d_in[20]; const float* lnLb = (const float*)d_in[21];
  float* out = (float*)d_out;

  const size_t MD = (size_t)MTOK * D_EMB;
  char* p = (char*)d_ws;
  float*  X    = (float*)p;  p += MD * sizeof(float);   // activation [L,B,D]
  float*  Qf   = (float*)p;  p += MD * sizeof(float);   // Q raw
  float*  Kf   = (float*)p;  p += MD * sizeof(float);   // K raw / Wd out
  float*  X2   = (float*)p;  p += MD * sizeof(float);   // post-double-LN (residual)
  bf16_t* Xb   = (bf16_t*)p; p += MD * sizeof(bf16_t);  // bf16 activation (GEMM A)
  bf16_t* Qn   = (bf16_t*)p; p += MD * sizeof(bf16_t);  // Qn / reused as W1 out
  bf16_t* Kn   = (bf16_t*)p; p += MD * sizeof(bf16_t);
  bf16_t* Vb   = (bf16_t*)p; p += MD * sizeof(bf16_t);  // V projection (bf16 direct)
  bf16_t* Cb   = (bf16_t*)p; p += MD * sizeof(bf16_t);  // attention ctx
  bf16_t* X2b  = (bf16_t*)p; p += MD * sizeof(bf16_t);  // bf16 double-LN out
  bf16_t* wts  = (bf16_t*)p;                            // 12 * 65536 bf16

  const size_t WSZ = (size_t)D_EMB * D_EMB;             // 65536
  const dim3 ggrid(MTOK / 256, 4);

  // Transpose+convert all 12 weight matrices to bf16 [N,K]
  const float* wsrc[12] = { Wq, Wk, Wv, Wd, W1, W2,
                            Wq + WSZ, Wk + WSZ, Wv + WSZ, Wd + WSZ, W1 + WSZ, W2 + WSZ };
  for (int i = 0; i < 12; ++i)
    prep_w<<<256, 256, 0, stream>>>(wsrc[i], wts + (size_t)i * WSZ);

  // Embedding (+mask), f32 + bf16
  embed_kernel<<<MTOK, 256, 0, stream>>>(seq, pos, item_emb, pos_emb, X, Xb);

  for (int i = 0; i < 2; ++i) {
    const bf16_t* Wtq = wts + (size_t)(i * 6 + 0) * WSZ;
    const bf16_t* Wtk = wts + (size_t)(i * 6 + 1) * WSZ;
    const bf16_t* Wtv = wts + (size_t)(i * 6 + 2) * WSZ;
    const bf16_t* Wtd = wts + (size_t)(i * 6 + 3) * WSZ;
    const bf16_t* Wt1 = wts + (size_t)(i * 6 + 4) * WSZ;
    const bf16_t* Wt2 = wts + (size_t)(i * 6 + 5) * WSZ;

    // Q,K raw f32; V straight to bf16
    gemm256<0><<<ggrid, 512, 0, stream>>>(Xb, Wtq, bq + i * 256, nullptr, nullptr, Qf, nullptr);
    gemm256<0><<<ggrid, 512, 0, stream>>>(Xb, Wtk, bk + i * 256, nullptr, nullptr, Kf, nullptr);
    gemm256<4><<<ggrid, 512, 0, stream>>>(Xb, Wtv, bv + i * 256, nullptr, nullptr, nullptr, Vb);

    // ELU + per-head L2 norm -> bf16
    elunorm_kernel<<<MTOK, 256, 0, stream>>>(Qf, Kf, Qn, Kn);

    // Linear attention -> ctx (bf16)
    attn_kernel<<<SEQLEN * NHEAD, 512, 0, stream>>>(Qn, Kn, Vb, Cb);

    // ctx @ Wd + bd + residual(X) -> Kf (f32)
    gemm256<1><<<ggrid, 512, 0, stream>>>(Cb, Wtd, bd + i * 256, X, nullptr, Kf, nullptr);

    // LN(.,1e-12) then LN(.,1e-8) -> X2 (f32) + X2b (bf16)
    double_ln_kernel<<<MTOK / 8, 256, 0, stream>>>(Kf, ln1g + i * 256, ln1b + i * 256,
                                                   lnfg + i * 256, lnfb + i * 256, X2, X2b);

    // PWFF: relu(X2@W1+b1) -> Qn (bf16, reuse) ; @W2+b2 + X2, mask -> X + Xb
    gemm256<2><<<ggrid, 512, 0, stream>>>(X2b, Wt1, b1 + i * 256, nullptr, nullptr, nullptr, Qn);
    gemm256<3><<<ggrid, 512, 0, stream>>>(Qn, Wt2, b2 + i * 256, X2, seq, X, Xb);
  }

  // Final LN + transpose to [B,L,D]
  final_ln_kernel<<<MTOK / 8, 256, 0, stream>>>(X, lnLg, lnLb, out);
}
